// GatedCanonLinearAttention_15487652069722
// MI455X (gfx1250) — matrix operand order repaired and checked by reading, not compiled, not run on hardware
//
#include <hip/hip_runtime.h>
#include <hip/hip_bf16.h>

// ---------------------------------------------------------------------------
// GLA (gated linear attention) for MI455X / gfx1250, wave32 + WMMA bf16.
// Pipeline:
//   (1) convert x and weights fp32->bf16
//   (2) bf16 WMMA GEMMs: q,k,v (bf16 out), gate g (fp32 out)
//   (3) low-rank gate decay: gk = logsigmoid((x@Wgk1)@Wgk2 + b)/16  (fp32)
//   (4) chunked WMMA scan per (b,h): state S 128x256 fp32 in LDS (+ bf16
//       transposed mirror for WMMA B-operand reads), chunk=32
//       -> per-head RMSNorm + SiLU(gate) fused, writes o bf16
//   (5) bf16 WMMA GEMM: out = o @ Wo  (fp32 out)
// Data movement: layout-preserving global->LDS staging uses gfx1250 async
// LDS loads (ASYNCcnt) when the builtin is available; transposing copies use
// the vmem+ds path.
// ---------------------------------------------------------------------------

#define B_    2
#define T_    2048
#define HID_  2048
#define H_    8
#define KD_   1024
#define VD_   2048
#define DK_   128
#define DV_   256
#define RANK_ 16
#define CHUNK 32
#define NCHUNK (T_ / CHUNK)
#define ROWS_ (B_ * T_)            // 4096
#define SCALE_ 0.08838834764831845f // DK^-0.5
#define EPS_   1e-5f

typedef unsigned short u16;
typedef __attribute__((ext_vector_type(16))) __bf16        v16bf;
typedef __attribute__((ext_vector_type(8)))  float         v8f;
typedef __attribute__((ext_vector_type(8)))  unsigned int  v8u;

// ---- gfx1250 async global->LDS support (guarded; fallback = sync copy) ----
#if defined(__has_builtin)
#  if __has_builtin(__builtin_amdgcn_global_load_async_to_lds_b128)
#    define ASYNC_LDS 1
#  endif
#endif
#ifndef ASYNC_LDS
#  define ASYNC_LDS 0
#endif

#if ASYNC_LDS
typedef int v4i_ __attribute__((vector_size(16)));
#define LDS_AS __attribute__((address_space(3)))
#define GLB_AS __attribute__((address_space(1)))
__device__ __forceinline__ void async_ld_b128(const void* g, void* l) {
  __builtin_amdgcn_global_load_async_to_lds_b128((GLB_AS v4i_*)g, (LDS_AS v4i_*)l, 0, 0);
}
__device__ __forceinline__ void wait_async0() {
#if __has_builtin(__builtin_amdgcn_s_wait_asynccnt)
  __builtin_amdgcn_s_wait_asynccnt(0);
#else
  asm volatile("s_wait_asynccnt 0x0" ::: "memory");
#endif
}
#endif

__device__ __forceinline__ u16 f2bf(float f) {
  unsigned int u = __float_as_uint(f);
  u += 0x7fffu + ((u >> 16) & 1u);           // RNE
  return (u16)(u >> 16);
}
__device__ __forceinline__ float bf2f(u16 h) {
  return __uint_as_float(((unsigned int)h) << 16);
}

__device__ __forceinline__ v8f wmma_bf16(v16bf a, v16bf b, v8f c) {
  return __builtin_amdgcn_wmma_f32_16x16x32_bf16(false, a, false, b,
                                                 (short)0, c, false, false);
}

// A fragment (16x32 bf16, M x K) from row-major LDS [rows][ld] (ld % 8 == 0).
// Layout (ISA 7.12.2): lanes 0-15: M=lane, K in {h8..h8+7, 16+h8..}, h8 = 0;
// lanes 16-31: M=lane-16, h8 = 8. Two 16B loads per lane.
__device__ __forceinline__ v16bf load_frag_a(const u16* base, int ld, int row0, int k0) {
  const int lane = threadIdx.x & 31;
  const int m = lane & 15;
  const int h8 = (lane >> 4) << 3;
  const u16* p = base + (row0 + m) * ld + k0 + h8;
  uint4 a = *(const uint4*)p;        // K: h8 .. h8+7
  uint4 b = *(const uint4*)(p + 16); // K: 16+h8 .. 16+h8+7
  v8u r = {a.x, a.y, a.z, a.w, b.x, b.y, b.z, b.w};
  return __builtin_bit_cast(v16bf, r);
}

// B fragment (32x16 bf16, K x N) from transposed LDS bt[n][k] (ld % 8 == 0).
// Layout: lanes 0-15: N=lane, K in {h8..h8+7, 16+h8..}, h8 = 0; lanes 16-31: N=lane-16, h8 = 8.
__device__ __forceinline__ v16bf load_frag_b(const u16* bt, int ld, int n0, int k0) {
  const int lane = threadIdx.x & 31;
  const int n = lane & 15;
  const int h16 = (lane >> 4) << 3;
  const u16* p = bt + (n0 + n) * ld + k0 + h16;
  uint4 a = *(const uint4*)p;        // K: h8 .. h8+7
  uint4 b = *(const uint4*)(p + 16); // K: 16+h8 .. 16+h8+7
  v8u r = {a.x, a.y, a.z, a.w, b.x, b.y, b.z, b.w};
  return __builtin_bit_cast(v16bf, r);
}

// ---------------------------------------------------------------------------
__global__ void cvt_kernel(const float* __restrict__ in, u16* __restrict__ out, int n) {
  int i = blockIdx.x * 256 + threadIdx.x;
  if (i < n) out[i] = f2bf(in[i]);
}

// tmp[r][j] = sum_k x[r][k] * Wgk1[k][j]   (4096 x 16, K=2048)
__global__ __launch_bounds__(256) void gk1_kernel(const float* __restrict__ x,
                                                  const float* __restrict__ w1,
                                                  float* __restrict__ tmp) {
  int i = blockIdx.x * 256 + threadIdx.x;
  int r = i >> 4, j = i & 15;
  const float* xr = x + (size_t)r * HID_;
  float acc = 0.f;
  for (int k = 0; k < HID_; k++) acc = fmaf(xr[k], w1[k * RANK_ + j], acc);
  tmp[(size_t)r * RANK_ + j] = acc;
}

// gkf[r][n] = logsigmoid(tmp[r][:] @ Wgk2[:,n] + bgk[n]) / 16
__global__ __launch_bounds__(256) void gk2_kernel(const float* __restrict__ tmp,
                                                  const float* __restrict__ w2,
                                                  const float* __restrict__ bgk,
                                                  float* __restrict__ gkf) {
  int i = blockIdx.x * 256 + threadIdx.x;
  int r = i >> 10, n = i & 1023;
  float acc = bgk[n];
#pragma unroll
  for (int k = 0; k < RANK_; k++) acc = fmaf(tmp[(size_t)r * RANK_ + k], w2[k * KD_ + n], acc);
  float ls = fminf(acc, 0.f) - log1pf(__expf(-fabsf(acc)));
  gkf[(size_t)i] = ls * (1.f / 16.f);
}

// ---------------------------------------------------------------------------
// Generic bf16 WMMA GEMM: C[M,N] = A[M,K] @ W[K,N], all row-major.
// Workgroup = 256 threads = 8 waves, 64x64 output tile, K step 32.
// A tile staged via async global->LDS (layout-preserving); W tile is
// transposed in flight (vmem load + ds stores) so B fragments are contiguous.
template <int OUTF32>
__global__ __launch_bounds__(256) void gemm_bf16_kernel(const u16* __restrict__ A,
                                                        const u16* __restrict__ W,
                                                        void* __restrict__ Cout,
                                                        int M, int N, int K) {
  __shared__ u16 aT[64 * 40];   // A tile, row-major [64][32], ld 40
  __shared__ u16 bT[64 * 40];   // W tile transposed [n][k], ld 40
  const int tid = threadIdx.x;
  const int lane = tid & 31;
  const int wave = tid >> 5;
  const int row0 = blockIdx.y * 64;
  const int col0 = blockIdx.x * 64;
  const int mi = wave >> 1;
  const int ni0 = (wave & 1) * 2;
  v8f acc0 = {0, 0, 0, 0, 0, 0, 0, 0};
  v8f acc1 = {0, 0, 0, 0, 0, 0, 0, 0};

  for (int kb = 0; kb < K; kb += 32) {
#if ASYNC_LDS
    {   // A: 64 rows x 4 x 16B, exactly one async b128 per thread
      int row = tid >> 2, seg = tid & 3;
      const u16* g = A + (size_t)(row0 + row) * K + kb + seg * 8;
      async_ld_b128(g, aT + row * 40 + seg * 8);
    }
#else
    for (int i = tid; i < 1024; i += 256) {          // A: 64 rows x 16 uint
      int row = i >> 4, cp = i & 15;
      unsigned int v = *(const unsigned int*)(A + (size_t)(row0 + row) * K + kb + cp * 2);
      *(unsigned int*)(aT + row * 40 + cp * 2) = v;
    }
#endif
    for (int i = tid; i < 1024; i += 256) {          // W: 32 k-rows x 32 uint, transpose
      int k = i >> 5, np = i & 31;
      unsigned int v = *(const unsigned int*)(W + (size_t)(kb + k) * N + col0 + np * 2);
      bT[(np * 2) * 40 + k]     = (u16)(v & 0xffffu);
      bT[(np * 2 + 1) * 40 + k] = (u16)(v >> 16);
    }
    if (kb + 32 < K) {   // cover L2 latency for the next K tile
      __builtin_prefetch(A + (size_t)(row0 + (tid >> 2)) * K + kb + 32, 0, 0);
      __builtin_prefetch(W + (size_t)(kb + 32 + (tid >> 3)) * N + col0, 0, 0);
    }
#if ASYNC_LDS
    wait_async0();
#endif
    __syncthreads();
    v16bf af = load_frag_a(aT, 40, mi * 16, 0);
    v16bf b0 = load_frag_b(bT, 40, ni0 * 16, 0);
    v16bf b1 = load_frag_b(bT, 40, (ni0 + 1) * 16, 0);
    acc0 = wmma_bf16(af, b0, acc0);
    acc1 = wmma_bf16(af, b1, acc1);
    __syncthreads();
  }

  const int n = lane & 15, half = lane >> 4;
#pragma unroll
  for (int r = 0; r < 8; r++) {
    int row = row0 + mi * 16 + r + 8 * half;   // C/D layout: M = r + 8*half, N = lane&15
    int c0 = col0 + ni0 * 16 + n;
    if (OUTF32) {
      ((float*)Cout)[(size_t)row * N + c0]      = acc0[r];
      ((float*)Cout)[(size_t)row * N + c0 + 16] = acc1[r];
    } else {
      ((u16*)Cout)[(size_t)row * N + c0]      = f2bf(acc0[r]);
      ((u16*)Cout)[(size_t)row * N + c0 + 16] = f2bf(acc1[r]);
    }
  }
}

// ---------------------------------------------------------------------------
// Chunked gated scan. One workgroup (8 waves) per (b,h) chain.
// Dynamic LDS layout (301,056 B, within CDNA5's 320 KB/workgroup):
//   S[128][256] f32, Obuf[32][256] f32, gcum[32][128] f32, gtot[128] f32,
//   SbT[256][136] bf16 (S transposed mirror, [n][k]), qs[32][136], kiv[32][136],
//   kdT[128][40] ([k][t]), vT[256][40] ([dv][t]), Ab[32][40] bf16.
__global__ __launch_bounds__(256) void gla_scan_kernel(const u16* __restrict__ qg,
                                                       const u16* __restrict__ kg,
                                                       const u16* __restrict__ vg,
                                                       const float* __restrict__ gkf,
                                                       const float* __restrict__ gfl,
                                                       const float* __restrict__ gnw,
                                                       u16* __restrict__ og) {
  extern __shared__ char smem_raw[];
  float* S    = (float*)smem_raw;            // 131072 B
  float* Obuf = S + 128 * 256;               //  32768 B
  float* gcum = Obuf + 32 * 256;             //  16384 B
  float* gtot = gcum + 32 * 128;             //    512 B
  u16* SbT = (u16*)(gtot + 128);             //  69632 B
  u16* qs  = SbT + 256 * 136;                //   8704 B
  u16* kiv = qs + 32 * 136;                  //   8704 B
  u16* kdT = kiv + 32 * 136;                 //  10240 B
  u16* vT  = kdT + 128 * 40;                 //  20480 B
  u16* Ab  = vT + 256 * 40;                  //   2560 B

  const int tid = threadIdx.x, lane = tid & 31, wave = tid >> 5;
  const int b = blockIdx.x >> 3, h = blockIdx.x & 7;
  const size_t qbase = (size_t)b * T_ * KD_ + (size_t)h * DK_;
  const size_t vbase = (size_t)b * T_ * VD_ + (size_t)h * DV_;

  for (int i = tid; i < 128 * 256; i += 256) S[i] = 0.f;
  for (int i = tid; i < 256 * 136; i += 256) SbT[i] = 0;
  __syncthreads();

  for (int ci = 0; ci < NCHUNK; ci++) {
    const int t0 = ci * CHUNK;

    // --- phase 1: load gk (async direct-to-LDS), prefix over t, stage ---
#if ASYNC_LDS
    for (int i = tid; i < 1024; i += 256) {  // 32 rows x 32 x 16B segments
      int t = i >> 5, seg = i & 31;
      const float* g = gkf + qbase + (size_t)(t0 + t) * KD_ + seg * 4;
      async_ld_b128(g, gcum + t * 128 + seg * 4);
    }
    wait_async0();
#else
    for (int i = tid; i < CHUNK * DK_; i += 256) {
      int t = i >> 7, k = i & 127;
      gcum[i] = gkf[qbase + (size_t)(t0 + t) * KD_ + k];
    }
#endif
    __syncthreads();
    if (tid < DK_) {
      float a = 0.f;
      for (int t = 0; t < CHUNK; t++) { a += gcum[t * DK_ + tid]; gcum[t * DK_ + tid] = a; }
      gtot[tid] = a;
    }
    __syncthreads();
    for (int i = tid; i < CHUNK * DK_; i += 256) {
      int t = i >> 7, k = i & 127;
      float g = gcum[i];
      float qv = bf2f(qg[qbase + (size_t)(t0 + t) * KD_ + k]);
      float kv = bf2f(kg[qbase + (size_t)(t0 + t) * KD_ + k]);
      qs[t * 136 + k]  = f2bf(qv * __expf(g) * SCALE_);       // q~ (scale folded)
      kiv[t * 136 + k] = f2bf(kv * __expf(-g));               // k~ for intra
      kdT[k * 40 + t]  = f2bf(kv * __expf(gtot[k] - g));      // k^ for state update
    }
    for (int i = tid; i < CHUNK * DV_; i += 256) {
      int t = i >> 8, d = i & 255;
      vT[d * 40 + t] = vg[vbase + (size_t)(t0 + t) * VD_ + d];
    }
    __syncthreads();

    // --- phase 2: A = q~ @ k~^T  (32x32, K=128), causal mask ---
    if (wave < 4) {
      int mi = wave >> 1, ni = wave & 1;
      v8f acc = {0, 0, 0, 0, 0, 0, 0, 0};
#pragma unroll
      for (int kk = 0; kk < 4; kk++) {
        v16bf af  = load_frag_a(qs, 136, mi * 16, kk * 32);
        v16bf bfr = load_frag_b(kiv, 136, ni * 16, kk * 32);
        acc = wmma_bf16(af, bfr, acc);
      }
      int n = lane & 15, half = lane >> 4;
#pragma unroll
      for (int r = 0; r < 8; r++) {
        int tl = mi * 16 + r + 8 * half;
        int sl = ni * 16 + n;
        Ab[tl * 40 + sl] = f2bf(sl <= tl ? acc[r] : 0.f);
      }
    }
    __syncthreads();

    // --- phase 3: O = q~ @ S_old + A_mask @ V  (32x256), 4 tiles/wave ---
#pragma unroll
    for (int i = 0; i < 4; i++) {
      int tile = wave * 4 + i, mi = tile >> 4, ni = tile & 15;
      v8f acc = {0, 0, 0, 0, 0, 0, 0, 0};
#pragma unroll
      for (int kk = 0; kk < 4; kk++) {
        v16bf af  = load_frag_a(qs, 136, mi * 16, kk * 32);
        v16bf bfr = load_frag_b(SbT, 136, ni * 16, kk * 32);
        acc = wmma_bf16(af, bfr, acc);
      }
      v16bf af2  = load_frag_a(Ab, 40, mi * 16, 0);
      v16bf bfr2 = load_frag_b(vT, 40, ni * 16, 0);
      acc = wmma_bf16(af2, bfr2, acc);
      int n = lane & 15, half = lane >> 4;
#pragma unroll
      for (int r = 0; r < 8; r++)
        Obuf[(mi * 16 + r + 8 * half) * DV_ + ni * 16 + n] = acc[r];
    }
    __syncthreads();

    // --- phase 4: S = diag(e^Gtot) S + k^T @ V ; wave owns K-slab = wave*16 ---
    {
      int n = lane & 15, half = lane >> 4;
      v16bf af = load_frag_a(kdT, 40, wave * 16, 0);
      for (int ni = 0; ni < 16; ni++) {
        v8f c = {0, 0, 0, 0, 0, 0, 0, 0};
#pragma unroll
        for (int r = 0; r < 8; r++) {
          int k = wave * 16 + r + 8 * half;
          c[r] = S[k * DV_ + ni * 16 + n] * __expf(gtot[k]);
        }
        v16bf bfr = load_frag_b(vT, 40, ni * 16, 0);
        v8f d = wmma_bf16(af, bfr, c);
#pragma unroll
        for (int r = 0; r < 8; r++) {
          int k = wave * 16 + r + 8 * half;
          int nn = ni * 16 + n;
          S[k * DV_ + nn] = d[r];
          SbT[nn * 136 + k] = f2bf(d[r]);
        }
      }
    }

    // --- phase 5: RMSNorm (per row over DV) + SiLU gate, store bf16 ---
    for (int rr = 0; rr < 4; rr++) {
      int t = wave * 4 + rr;
      float ss = 0.f;
#pragma unroll
      for (int j = 0; j < 8; j++) {
        float v = Obuf[t * DV_ + lane + 32 * j];
        ss += v * v;
      }
#pragma unroll
      for (int off = 16; off >= 1; off >>= 1) ss += __shfl_xor(ss, off, 32);
      float rinv = rsqrtf(ss * (1.f / DV_) + EPS_);
      size_t rowoff = vbase + (size_t)(t0 + t) * VD_;
#pragma unroll
      for (int j = 0; j < 8; j++) {
        int dv = lane + 32 * j;
        float o = Obuf[t * DV_ + dv] * rinv * gnw[dv];
        float gv = gfl[rowoff + dv];
        o *= gv / (1.f + __expf(-gv));     // * silu(g)
        og[rowoff + dv] = f2bf(o);
      }
    }
    __syncthreads();
  }
}

// ---------------------------------------------------------------------------
extern "C" void kernel_launch(void* const* d_in, const int* in_sizes, int n_in,
                              void* d_out, int out_size, void* d_ws, size_t ws_size,
                              hipStream_t stream) {
  const float* x    = (const float*)d_in[0];
  const float* Wq   = (const float*)d_in[1];
  const float* Wk   = (const float*)d_in[2];
  const float* Wv   = (const float*)d_in[3];
  const float* Wg   = (const float*)d_in[4];
  const float* Wgk1 = (const float*)d_in[5];
  const float* Wgk2 = (const float*)d_in[6];
  const float* bgk  = (const float*)d_in[7];
  const float* gnw  = (const float*)d_in[8];
  const float* Wo   = (const float*)d_in[9];
  (void)in_sizes; (void)n_in; (void)out_size; (void)ws_size;

  char* ws = (char*)d_ws;
  // workspace layout (bytes)
  u16*   xb   = (u16*)(ws + 0);            // 4096*2048 bf16  = 16,777,216
  u16*   wqb  = (u16*)(ws + 16777216);     // 2048*1024       =  4,194,304
  u16*   wkb  = (u16*)(ws + 20971520);     //                  4,194,304
  u16*   wvb  = (u16*)(ws + 25165824);     // 2048*2048       =  8,388,608
  u16*   wgb  = (u16*)(ws + 33554432);     //                  8,388,608
  u16*   wob  = (u16*)(ws + 41943040);     //                  8,388,608
  u16*   qb   = (u16*)(ws + 50331648);     // 4096*1024       =  8,388,608
  u16*   kb   = (u16*)(ws + 58720256);     //                  8,388,608
  u16*   vb   = (u16*)(ws + 67108864);     // 4096*2048       = 16,777,216
  u16*   ob   = (u16*)(ws + 83886080);     // 4096*2048       = 16,777,216
  float* tmp16= (float*)(ws + 100663296);  // 4096*16 f32     =    262,144
  float* gkf  = (float*)(ws + 100925440);  // 4096*1024 f32   = 16,777,216
  float* gfl  = (float*)(ws + 117702656);  // 4096*2048 f32   = 33,554,432
  // end: 151,257,088 bytes

  // (1) fp32 -> bf16 conversions
  cvt_kernel<<<(ROWS_ * HID_ + 255) / 256, 256, 0, stream>>>(x, xb, ROWS_ * HID_);
  cvt_kernel<<<(HID_ * KD_ + 255) / 256, 256, 0, stream>>>(Wq, wqb, HID_ * KD_);
  cvt_kernel<<<(HID_ * KD_ + 255) / 256, 256, 0, stream>>>(Wk, wkb, HID_ * KD_);
  cvt_kernel<<<(HID_ * VD_ + 255) / 256, 256, 0, stream>>>(Wv, wvb, HID_ * VD_);
  cvt_kernel<<<(HID_ * VD_ + 255) / 256, 256, 0, stream>>>(Wg, wgb, HID_ * VD_);
  cvt_kernel<<<(VD_ * HID_ + 255) / 256, 256, 0, stream>>>(Wo, wob, VD_ * HID_);

  // (2) low-rank gate decay path (fp32)
  gk1_kernel<<<ROWS_ * RANK_ / 256, 256, 0, stream>>>(x, Wgk1, tmp16);
  gk2_kernel<<<ROWS_ * KD_ / 256, 256, 0, stream>>>(tmp16, Wgk2, bgk, gkf);

  // (3) projection GEMMs (bf16 WMMA)
  gemm_bf16_kernel<0><<<dim3(KD_ / 64, ROWS_ / 64), 256, 0, stream>>>(xb, wqb, qb, ROWS_, KD_, HID_);
  gemm_bf16_kernel<0><<<dim3(KD_ / 64, ROWS_ / 64), 256, 0, stream>>>(xb, wkb, kb, ROWS_, KD_, HID_);
  gemm_bf16_kernel<0><<<dim3(VD_ / 64, ROWS_ / 64), 256, 0, stream>>>(xb, wvb, vb, ROWS_, VD_, HID_);
  gemm_bf16_kernel<1><<<dim3(VD_ / 64, ROWS_ / 64), 256, 0, stream>>>(xb, wgb, gfl, ROWS_, VD_, HID_);

  // (4) chunked WMMA scan + RMSNorm + SiLU gate
  const size_t smem = 301056;
  gla_scan_kernel<<<B_ * H_, 256, smem, stream>>>(qb, kb, vb, gkf, gfl, gnw, ob);

  // (5) output projection: d_out = o @ Wo (fp32 out)
  gemm_bf16_kernel<1><<<dim3(HID_ / 64, ROWS_ / 64), 256, 0, stream>>>(ob, wob, d_out, ROWS_, HID_, VD_);
}
